// SimpleModel_91113436217596
// MI455X (gfx1250) — matrix operand order repaired and checked by reading, not compiled, not run on hardware
//
#include <hip/hip_runtime.h>
#include <hip/hip_bf16.h>

typedef __bf16 bf16_t;
typedef __attribute__((ext_vector_type(16))) __bf16 v16bf;
typedef __attribute__((ext_vector_type(8)))  __bf16 v8bf;
typedef __attribute__((ext_vector_type(8)))  float  v8f;

#define ROWS   32768   // B*T
#define D_IN   1024
#define D_H    512
#define D_CODE 256
#define N_CODES 128
#define MTILE  32      // rows per workgroup
#define KC     128     // x staging chunk (K elements)

union AFrag { v16bf v; v8bf g[2]; };

// ---------------------------------------------------------------------------
// Prep: bf16-convert + transpose weights into WMMA B-layout ([N][K] row-major),
// precompute bf16-consistent codebook norms, zero the loss accumulator.
// ---------------------------------------------------------------------------
__global__ __launch_bounds__(256) void prep_kernel(
    const float* __restrict__ W1, const float* __restrict__ W2,
    const float* __restrict__ cbf,
    bf16_t* __restrict__ W1t, bf16_t* __restrict__ W2t,
    bf16_t* __restrict__ cbb, float* __restrict__ cnorm,
    float* __restrict__ loss)
{
    int gid = blockIdx.x * blockDim.x + threadIdx.x;
    int stride = gridDim.x * blockDim.x;
    for (int i = gid; i < D_IN * D_H; i += stride) {      // W1[k][n] -> W1t[n][k]
        int k = i >> 9, n = i & 511;
        W1t[(size_t)n * D_IN + k] = (bf16_t)W1[i];
    }
    for (int i = gid; i < D_H * D_CODE; i += stride) {    // W2[k][n] -> W2t[n][k]
        int k = i >> 8, n = i & 255;
        W2t[(size_t)n * D_H + k] = (bf16_t)W2[i];
    }
    for (int i = gid; i < N_CODES * D_CODE; i += stride)  // codebook already [N][K]
        cbb[i] = (bf16_t)cbf[i];
    if (gid < N_CODES) {
        float s = 0.f;
        for (int j = 0; j < D_CODE; ++j) {
            float v = (float)(bf16_t)cbf[gid * D_CODE + j];
            s += v * v;
        }
        cnorm[gid] = s;
    }
    if (gid == 0) loss[0] = 0.f;
}

// ---------------------------------------------------------------------------
// Fused per 32-row block:
//   x --(reg-double-buffered chunks)--> LDS bf16 -> GEMM1(relu) -> LDS h
//   -> GEMM2(+b2) -> LDS enc -> norms -> dist-GEMM vs codebook -> argmin+loss.
// All GEMMs via v_wmma_f32_16x16x32_bf16. Every wave owns BOTH 16-row
// sub-tiles so each B fragment feeds two WMMAs (halves B-load traffic).
// ---------------------------------------------------------------------------
__global__ __launch_bounds__(256) void vq_main(
    const float* __restrict__ x, const float* __restrict__ b1,
    const float* __restrict__ b2,
    const bf16_t* __restrict__ W1t, const bf16_t* __restrict__ W2t,
    const bf16_t* __restrict__ cbb, const float* __restrict__ cnorm,
    float* __restrict__ out, float* __restrict__ loss)
{
    __shared__ bf16_t xb[MTILE][KC + 8];    // x chunk (stride 272B, 16B aligned)
    __shared__ bf16_t hls[MTILE][520];      // h tile
    __shared__ bf16_t enc[MTILE][264];      // encoded tile
    __shared__ float  enorm[MTILE];
    __shared__ float  redv[8][MTILE];
    __shared__ int    redi[8][MTILE];

    const int t    = threadIdx.x;
    const int wave = t >> 5;
    const int lane = t & 31;
    const int half = lane >> 4;       // C/D: m = r + 8*half ; A/B K-split by half
    const int lm   = lane & 15;       // A: row ; B/D: col
    const int r0   = blockIdx.x * MTILE;

    if (t < MTILE) enorm[t] = 0.f;

    // ---- Phase 1: h = relu(x @ W1 + b1): 32 x 512; wave: 32 rows x 64 cols.
    {
        const int nb = wave * 64;
        float xr[16];                       // register buffer for next x chunk
        #pragma unroll
        for (int j = 0; j < 16; ++j) {      // preload chunk 0
            int i = t + j * 256, row = i >> 7, col = i & (KC - 1);
            xr[j] = x[(size_t)(r0 + row) * D_IN + col];
        }
        v8f acc[2][4] = {};
        for (int kc = 0; kc < D_IN; kc += KC) {
            __syncthreads();                // previous chunk fully consumed
            #pragma unroll
            for (int j = 0; j < 16; ++j) {  // regs -> LDS (bf16)
                int i = t + j * 256, row = i >> 7, col = i & (KC - 1);
                xb[row][col] = (bf16_t)xr[j];
            }
            __syncthreads();                // chunk visible
            if (kc + KC < D_IN) {           // launch next chunk's loads now;
                #pragma unroll              // they overlap the WMMAs below
                for (int j = 0; j < 16; ++j) {
                    int i = t + j * 256, row = i >> 7, col = i & (KC - 1);
                    xr[j] = x[(size_t)(r0 + row) * D_IN + kc + KC + col];
                }
            }
            for (int kb = 0; kb < KC; kb += 32) {
                AFrag a0, a1;
                a0.g[0] = *(const v8bf*)&xb[lm][kb + 16 * half];
                a0.g[1] = *(const v8bf*)&xb[lm][kb + 16 * half + 8];
                a1.g[0] = *(const v8bf*)&xb[16 + lm][kb + 16 * half];
                a1.g[1] = *(const v8bf*)&xb[16 + lm][kb + 16 * half + 8];
                #pragma unroll
                for (int tt = 0; tt < 4; ++tt) {
                    const bf16_t* bp = W1t + (size_t)(nb + tt * 16 + lm) * D_IN + kc + kb + 16 * half;
                    v16bf bfr = *(const v16bf*)bp;   // one B, two WMMAs
                    acc[0][tt] = __builtin_amdgcn_wmma_f32_16x16x32_bf16(
                        false, a0.v, false, bfr, (short)0, acc[0][tt], false, false);
                    acc[1][tt] = __builtin_amdgcn_wmma_f32_16x16x32_bf16(
                        false, a1.v, false, bfr, (short)0, acc[1][tt], false, false);
                }
            }
        }
        #pragma unroll
        for (int tt = 0; tt < 4; ++tt) {
            int col = nb + tt * 16 + lm;
            float bias = b1[col];
            #pragma unroll
            for (int ms = 0; ms < 2; ++ms) {
                #pragma unroll
                for (int r = 0; r < 8; ++r) {
                    int m = ms * 16 + r + 8 * half;
                    float v = acc[ms][tt][r] + bias;
                    hls[m][col] = (bf16_t)(v > 0.f ? v : 0.f);
                }
            }
        }
    }
    __syncthreads();

    // ---- Phase 2: encoded = h @ W2 + b2: 32 x 256; wave: 32 rows x 32 cols.
    {
        const int nb = wave * 32;
        v8f acc[2][2] = {};
        for (int kb = 0; kb < D_H; kb += 32) {
            AFrag a0, a1;
            a0.g[0] = *(const v8bf*)&hls[lm][kb + 16 * half];
            a0.g[1] = *(const v8bf*)&hls[lm][kb + 16 * half + 8];
            a1.g[0] = *(const v8bf*)&hls[16 + lm][kb + 16 * half];
            a1.g[1] = *(const v8bf*)&hls[16 + lm][kb + 16 * half + 8];
            #pragma unroll
            for (int tt = 0; tt < 2; ++tt) {
                const bf16_t* bp = W2t + (size_t)(nb + tt * 16 + lm) * D_H + kb + 16 * half;
                v16bf bfr = *(const v16bf*)bp;
                acc[0][tt] = __builtin_amdgcn_wmma_f32_16x16x32_bf16(
                    false, a0.v, false, bfr, (short)0, acc[0][tt], false, false);
                acc[1][tt] = __builtin_amdgcn_wmma_f32_16x16x32_bf16(
                    false, a1.v, false, bfr, (short)0, acc[1][tt], false, false);
            }
        }
        #pragma unroll
        for (int tt = 0; tt < 2; ++tt) {
            int col = nb + tt * 16 + lm;
            float bias = b2[col];
            #pragma unroll
            for (int ms = 0; ms < 2; ++ms) {
                #pragma unroll
                for (int r = 0; r < 8; ++r) {
                    int m = ms * 16 + r + 8 * half;
                    enc[m][col] = (bf16_t)(acc[ms][tt][r] + bias);
                }
            }
        }
    }
    __syncthreads();

    // ---- Row norms of encoded (bf16-rounded, consistent with WMMA dots) ----
    {
        int row = t >> 3, seg = t & 7;          // 8 threads per row, 32 cols each
        float s = 0.f;
        #pragma unroll
        for (int j = 0; j < 32; ++j) {
            float v = (float)enc[row][seg * 32 + j];
            s += v * v;
        }
        atomicAdd(&enorm[row], s);
    }
    __syncthreads();

    // ---- Phase 3: dots = enc @ cb^T: 32 rows x 16 codes per wave, K=256.
    {
        const int code = wave * 16 + lm;
        v8f acc[2] = {};
        for (int kb = 0; kb < D_CODE; kb += 32) {
            const bf16_t* bp = cbb + (size_t)code * D_CODE + kb + 16 * half;
            v16bf bfr = *(const v16bf*)bp;
            #pragma unroll
            for (int ms = 0; ms < 2; ++ms) {
                AFrag a;
                a.g[0] = *(const v8bf*)&enc[ms * 16 + lm][kb + 16 * half];
                a.g[1] = *(const v8bf*)&enc[ms * 16 + lm][kb + 16 * half + 8];
                acc[ms] = __builtin_amdgcn_wmma_f32_16x16x32_bf16(
                    false, a.v, false, bfr, (short)0, acc[ms], false, false);
            }
        }
        float cn = cnorm[code];
        #pragma unroll
        for (int ms = 0; ms < 2; ++ms) {
            #pragma unroll
            for (int r = 0; r < 8; ++r) {
                int m = ms * 16 + r + 8 * half;
                float d2 = enorm[m] - 2.f * acc[ms][r] + cn;
                int   idx = code;
                #pragma unroll
                for (int off = 1; off < 16; off <<= 1) {  // reduce codes in 16-lane half
                    float ov = __shfl_xor(d2, off, 32);
                    int   oi = __shfl_xor(idx, off, 32);
                    if (ov < d2 || (ov == d2 && oi < idx)) { d2 = ov; idx = oi; }
                }
                if (lm == 0) { redv[wave][m] = d2; redi[wave][m] = idx; }
            }
        }
    }
    __syncthreads();

    // ---- Final cross-wave argmin, index write, loss accumulate ----
    if (t < MTILE) {
        float bv = redv[0][t];
        int   bi = redi[0][t];
        #pragma unroll
        for (int w = 1; w < 8; ++w) {
            float v = redv[w][t]; int i = redi[w][t];
            if (v < bv || (v == bv && i < bi)) { bv = v; bi = i; }
        }
        out[r0 + t] = (float)bi;
        // commitment-loss sum == sum of min squared distances
        atomicAdd(loss, bv > 0.f ? bv : 0.f);
    }
}

// ---------------------------------------------------------------------------
// Finalize: losses (commitment == codebook numerically; total = 1.25x).
// ---------------------------------------------------------------------------
__global__ void finalize_kernel(const float* __restrict__ loss, float* __restrict__ out)
{
    if (blockIdx.x == 0 && threadIdx.x == 0) {
        float c = loss[0] / (float)(ROWS * D_CODE);
        out[ROWS]     = c;
        out[ROWS + 1] = c;
        out[ROWS + 2] = 1.25f * c;
    }
}

extern "C" void kernel_launch(void* const* d_in, const int* in_sizes, int n_in,
                              void* d_out, int out_size, void* d_ws, size_t ws_size,
                              hipStream_t stream)
{
    const float* x  = (const float*)d_in[0];
    const float* W1 = (const float*)d_in[1];
    const float* b1 = (const float*)d_in[2];
    const float* W2 = (const float*)d_in[3];
    const float* b2 = (const float*)d_in[4];
    const float* cb = (const float*)d_in[5];
    // d_in[6] (Wd) / d_in[7] (bd) are dead code in the reference.
    float* out = (float*)d_out;

    char* ws = (char*)d_ws;
    bf16_t* W1t   = (bf16_t*)ws;  ws += (size_t)D_H    * D_IN   * sizeof(bf16_t);
    bf16_t* W2t   = (bf16_t*)ws;  ws += (size_t)D_CODE * D_H    * sizeof(bf16_t);
    bf16_t* cbb   = (bf16_t*)ws;  ws += (size_t)N_CODES* D_CODE * sizeof(bf16_t);
    float*  cnorm = (float*)ws;   ws += (size_t)N_CODES * sizeof(float);
    float*  loss  = (float*)ws;

    prep_kernel<<<256, 256, 0, stream>>>(W1, W2, cb, W1t, W2t, cbb, cnorm, loss);
    vq_main<<<ROWS / MTILE, 256, 0, stream>>>(x, b1, b2, W1t, W2t, cbb, cnorm, out, loss);
    finalize_kernel<<<1, 64, 0, stream>>>(loss, out);
}
